// AttentionInteractionBlock_9028021256837
// MI455X (gfx1250) — matrix operand order repaired and checked by reading, not compiled, not run on hardware
//
#include <hip/hip_runtime.h>

// Problem constants (from reference)
#define Nn   50000
#define Ee   640000
#define NTILE (Nn/16)      // 3125 node tiles (exact)
#define ETILE (Ee/16)      // 40000 edge tiles (exact)

typedef __attribute__((ext_vector_type(16))) _Float16 v16h;
typedef __attribute__((ext_vector_type(8)))  float    v8f;
typedef _Float16 half_t;

// Blob offsets (in halfs). Each WMMA B-tile = 32 lanes * 16 halfs = 512 halfs.
#define OFF_KW   0        // 8 tiles (4 heads x 2 nsubs)
#define OFF_VW   4096     // 8 tiles
#define OFF_QT   8192     // 8 tiles (wkl^T @ q_w folded)
#define OFF_WK1  12288    // 4 tiles (2 nsubs x 2 kchunks)
#define OFF_WK2  14336    // 2 tiles
#define OFF_WV1  15360    // 4 tiles
#define OFF_WV2  17408    // 2 tiles
#define OFF_WVL  18432    // 2 tiles
#define OFF_OUT  19456    // 32 tiles (8 nsubs x 4 kchunks)
#define BLOB_HALFS 35840

__device__ __forceinline__ v8f wmma_f16(v16h a, v16h b, v8f c) {
  // v_wmma_f32_16x16x32_f16: (neg_a, A, neg_b, B, c_mod, C, reuse_a, reuse_b)
  return __builtin_amdgcn_wmma_f32_16x16x32_f16(false, a, false, b, (short)0, c, false, false);
}

// ---- CDNA5 async global->LDS copy (ASYNCcnt path, ISA 10. / 15.18.3) --------
// VDST = VGPR holding LDS byte address (low 32 bits of the flat shared ptr,
// per ISA 10.2 aperture mapping), VADDR = 64-bit global address VGPR pair.
__device__ __forceinline__ void async_cp_b128(float* lds_dst, const float* gsrc) {
  unsigned l = (unsigned)(uintptr_t)lds_dst;
  asm volatile("global_load_async_to_lds_b128 %0, %1, off"
               :: "v"(l), "v"(gsrc)
               : "memory");
}
__device__ __forceinline__ void async_wait() {
  asm volatile("s_wait_asynccnt 0x0" ::: "memory");
}

// A-matrix 16x32 f16 layout (ISA 7.12.2): lane holds row (lane&15);
// lanes 0-15: K = {0..7,16..23}; lanes 16-31: K = {8..15,24..31}.
__device__ __forceinline__ v16h load_a_lds(const float* base, int pitch, int lane) {
  v16h a; int r = lane & 15; int hi = (lane >> 4) << 3;
  #pragma unroll
  for (int i = 0; i < 16; ++i) {
    int k = hi + (i & 7) + ((i >> 3) << 4);
    a[i] = (half_t)base[r * pitch + k];
  }
  return a;
}
// B tile pre-laid-out in memory: lane-contiguous 16 halfs (32B) per lane.
__device__ __forceinline__ v16h load_b_blob(const half_t* tile, int lane) {
  return *(const v16h*)(tile + lane * 16);
}
__device__ __forceinline__ float sspf(float x) {  // softplus(x) - log(2)
  return fmaxf(x, 0.f) + log1pf(__expf(-fabsf(x))) - 0.69314718056f;
}
// Order-preserving float<->uint transform for atomicMax on floats.
__device__ __forceinline__ unsigned ordf(float f) {
  unsigned u = __float_as_uint(f);
  return (u & 0x80000000u) ? ~u : (u | 0x80000000u);
}
__device__ __forceinline__ float unordf(unsigned u) {
  return __uint_as_float((u & 0x80000000u) ? (u & 0x7fffffffu) : ~u);
}

// ---------------- prep: fold weights, convert to f16 B-layout blobs ----------
__device__ void fill_blob(half_t* dst, const float* W, int ldk, int nsubs, int kchunks, int tid) {
  int total = nsubs * kchunks * 512;
  for (int t = tid; t < total; t += 256) {
    int tile = t >> 9; int e = t & 511; int lane = e >> 4; int i = e & 15;
    int ns = tile / kchunks; int kc = tile % kchunks;
    int n = ns * 16 + (lane & 15);
    int k = kc * 32 + (((lane >> 4) & 1) << 3) + (i & 7) + ((i >> 3) << 4);   // B layout: K=((lane>>4)<<3)+(i&7)+((i>>3)<<4)
    dst[t] = (half_t)W[n * ldk + k];
  }
}

__global__ void prep_kernel(const float* k_w, const float* q_w, const float* v_w,
                            const float* wkl_w, const float* wkl_b,
                            const float* wk1_w, const float* wk2_w,
                            const float* wv1_w, const float* wv2_w,
                            const float* wvl_w, const float* out_w,
                            float* qtw, float* u_vec, half_t* blobs) {
  int tid = threadIdx.x;
  // M_h[k',d] = sum_k wkl_w[k,k'] * q_w[h,k,d]
  for (int t = tid; t < 4 * 32 * 32; t += 256) {
    int h = t >> 10; int kp = (t >> 5) & 31; int d = t & 31;
    float s = 0.f;
    for (int k = 0; k < 32; ++k) s += wkl_w[k * 32 + kp] * q_w[h * 1024 + k * 32 + d];
    qtw[t] = s;
  }
  // u[h,d] = sum_k q_w[h,k,d] * wkl_b[k]
  for (int t = tid; t < 128; t += 256) {
    int h = t >> 5; int d = t & 31;
    float s = 0.f;
    for (int k = 0; k < 32; ++k) s += q_w[h * 1024 + k * 32 + d] * wkl_b[k];
    u_vec[t] = s;
  }
  __syncthreads();
  for (int h = 0; h < 4; ++h) {
    fill_blob(blobs + OFF_KW + h * 1024, k_w + h * 1024, 32, 2, 1, tid);
    fill_blob(blobs + OFF_VW + h * 1024, v_w + h * 1024, 32, 2, 1, tid);
    fill_blob(blobs + OFF_QT + h * 1024, qtw + h * 1024, 32, 2, 1, tid);
  }
  fill_blob(blobs + OFF_WK1, wk1_w, 64, 2, 2, tid);
  fill_blob(blobs + OFF_WK2, wk2_w, 32, 2, 1, tid);
  fill_blob(blobs + OFF_WV1, wv1_w, 64, 2, 2, tid);
  fill_blob(blobs + OFF_WV2, wv2_w, 32, 2, 1, tid);
  fill_blob(blobs + OFF_WVL, wvl_w, 32, 2, 1, tid);
  fill_blob(blobs + OFF_OUT, out_w, 128, 8, 4, tid);
}

__global__ void init_kernel(unsigned* m_ord, float* denom) {
  int tid = blockIdx.x * blockDim.x + threadIdx.x;
  if (tid < Nn * 4) { m_ord[tid] = 0x007FFFFFu /* ord(-inf) */; denom[tid] = 0.f; }
}

// ---------------- async staging helpers ---------------------------------------
// 16 rows x 64 floats (edge_attr tile) -> LDS with pitch 68 (pad vs bank conflicts)
__device__ __forceinline__ void stage16x64_async(const float* gsrc, float* dst68, int lane) {
  #pragma unroll
  for (int rep = 0; rep < 8; ++rep) {
    int t = rep * 32 + lane;
    int r = t >> 4, c = (t & 15) << 2;
    async_cp_b128(dst68 + r * 68 + c, gsrc + t * 4);
  }
}
// 16 rows x 128 floats (node tile) -> LDS with pitch 132
__device__ __forceinline__ void stage16x128_async(const float* gsrc, float* dst132, int lane) {
  #pragma unroll
  for (int rep = 0; rep < 16; ++rep) {
    int t = rep * 32 + lane;
    int r = t >> 5, c = (t & 31) << 2;
    async_cp_b128(dst132 + r * 132 + c, gsrc + t * 4);
  }
}

// ---------------- node pass 1: LN1 + per-head WMMA projections ----------------
__device__ __forceinline__ void proj_store(v16h a, const half_t* blob, float* dst,
                                           size_t base, int h, int lane) {
  int m0 = (lane >> 4) << 3, ncol = lane & 15;
  #pragma unroll
  for (int ns = 0; ns < 2; ++ns) {
    v8f c;
    #pragma unroll
    for (int j = 0; j < 8; ++j) c[j] = 0.f;
    v16h b = load_b_blob(blob + ns * 512, lane);
    c = wmma_f16(a, b, c);
    #pragma unroll
    for (int j = 0; j < 8; ++j)
      dst[(base + (size_t)(j + m0)) * 128 + h * 32 + ns * 16 + ncol] = c[j];
  }
}

__global__ void node_pass1(const float* x, const float* ln1_g, const float* ln1_b,
                           const half_t* kwb, const half_t* vwb, const half_t* qtb,
                           const float* u_vec,
                           float* h_k, float* h_v, float* qt_arr, float* qb) {
  __shared__ float s_buf[4][16 * 132];
  __shared__ float s_mean[4][16], s_rstd[4][16];
  int wid = threadIdx.x >> 5, lane = threadIdx.x & 31;
  int tile = blockIdx.x * 4 + wid;
  if (tile >= NTILE) return;                 // wave-uniform
  float* buf = s_buf[wid];
  size_t base = (size_t)tile * 16;
  stage16x128_async(x + base * 128, buf, lane);
  async_wait();
  if (lane < 16) {
    float s = 0.f, s2 = 0.f;
    for (int c = 0; c < 128; ++c) { float v = buf[lane * 132 + c]; s += v; s2 += v * v; }
    float m = s * (1.f / 128.f);
    float var = s2 * (1.f / 128.f) - m * m;
    s_mean[wid][lane] = m; s_rstd[wid][lane] = rsqrtf(var + 1e-5f);
  }
  for (int t = lane; t < 2048; t += 32) {
    int r = t >> 7, c = t & 127;
    buf[r * 132 + c] = (buf[r * 132 + c] - s_mean[wid][r]) * s_rstd[wid][r] * ln1_g[c] + ln1_b[c];
  }
  for (int h = 0; h < 4; ++h) {
    v16h a = load_a_lds(buf + h * 32, 132, lane);
    proj_store(a, kwb + h * 1024, h_k,    base, h, lane);
    proj_store(a, vwb + h * 1024, h_v,    base, h, lane);
    proj_store(a, qtb + h * 1024, qt_arr, base, h, lane);
  }
  #pragma unroll
  for (int rep = 0; rep < 2; ++rep) {
    int idx = rep * 32 + lane; int r = idx & 15, h = idx >> 4;
    float s = 0.f;
    #pragma unroll
    for (int k = 0; k < 32; ++k) s += buf[r * 132 + h * 32 + k] * u_vec[h * 32 + k];
    qb[(base + r) * 4 + h] = s;
  }
}

// ---------------- edge MLP tile helper (64 -> ssp -> 32 via WMMA) -------------
__device__ __forceinline__ void edge_mlp_tile(const float* ea_s, float* hid_s, float* w_s,
                                              const half_t* b1, const half_t* b2,
                                              const float* bias1, const float* bias2, int lane) {
  int m0 = (lane >> 4) << 3, ncol = lane & 15;
  v16h a0 = load_a_lds(ea_s, 68, lane);
  v16h a1 = load_a_lds(ea_s + 32, 68, lane);
  #pragma unroll
  for (int ns = 0; ns < 2; ++ns) {
    v8f c;
    #pragma unroll
    for (int j = 0; j < 8; ++j) c[j] = 0.f;
    c = wmma_f16(a0, load_b_blob(b1 + (ns * 2 + 0) * 512, lane), c);
    c = wmma_f16(a1, load_b_blob(b1 + (ns * 2 + 1) * 512, lane), c);
    float bb = bias1[ns * 16 + ncol];
    #pragma unroll
    for (int j = 0; j < 8; ++j) hid_s[(j + m0) * 36 + ns * 16 + ncol] = sspf(c[j] + bb);
  }
  v16h a2 = load_a_lds(hid_s, 36, lane);
  #pragma unroll
  for (int ns = 0; ns < 2; ++ns) {
    v8f c;
    #pragma unroll
    for (int j = 0; j < 8; ++j) c[j] = 0.f;
    c = wmma_f16(a2, load_b_blob(b2 + ns * 512, lane), c);
    float bb = bias2[ns * 16 + ncol];
    #pragma unroll
    for (int j = 0; j < 8; ++j) w_s[(j + m0) * 36 + ns * 16 + ncol] = c[j] + bb;
  }
}

// ---------------- edge pass A: Wk MLP -> qk -> atomic max ---------------------
__global__ void edge_pass_a(const float* ea, const int* eidx,
                            const half_t* wk1b, const half_t* wk2b,
                            const float* wk1_bias, const float* wk2_bias,
                            const float* qt_arr, const float* h_k, const float* qb,
                            float* qk_ws, unsigned* m_ord) {
  __shared__ float s_ea[4][16 * 68];
  __shared__ float s_hid[4][16 * 36];
  __shared__ float s_w[4][16 * 36];
  int wid = threadIdx.x >> 5, lane = threadIdx.x & 31;
  int tile = blockIdx.x * 4 + wid;               // exact: 10000 blocks * 4
  size_t ebase = (size_t)tile * 16;
  stage16x64_async(ea + ebase * 64, s_ea[wid], lane);
  async_wait();
  edge_mlp_tile(s_ea[wid], s_hid[wid], s_w[wid], wk1b, wk2b, wk1_bias, wk2_bias, lane);
  const float* w_s = s_w[wid];
  #pragma unroll
  for (int rep = 0; rep < 2; ++rep) {
    int idx = rep * 32 + lane; int el = idx & 15, h = idx >> 4;
    size_t e = ebase + el;
    int r = eidx[e], cc = eidx[(size_t)Ee + e];
    const float* qp = qt_arr + (size_t)r * 128 + h * 32;
    const float* kp = h_k + (size_t)cc * 128 + h * 32;
    float s = qb[(size_t)r * 4 + h];
    #pragma unroll
    for (int k = 0; k < 32; ++k) s += qp[k] * w_s[el * 36 + k] * kp[k];
    qk_ws[e * 4 + h] = s;
    atomicMax(m_ord + (size_t)r * 4 + h, ordf(s));
  }
}

// ---------------- edge pass B: exp + denom ------------------------------------
__global__ void edge_pass_b(const int* eidx, const float* qk_ws, const unsigned* m_ord,
                            float* ex_ws, float* denom) {
  int tid = blockIdx.x * blockDim.x + threadIdx.x;
  if (tid >= Ee * 4) return;
  int e = tid >> 2, h = tid & 3;
  int r = eidx[e];
  float m = unordf(m_ord[r * 4 + h]);
  float ex = __expf(qk_ws[tid] - m);
  ex_ws[tid] = ex;
  atomicAdd(denom + (size_t)r * 4 + h, ex);
}

// ---------------- edge pass C: Wv MLP -> weighted scatter-add -----------------
__global__ void edge_pass_c(const float* ea, const int* eidx,
                            const half_t* wv1b, const half_t* wv2b,
                            const float* wv1_bias, const float* wv2_bias,
                            const float* h_v, const float* ex_ws, const float* denom,
                            float* t_acc) {
  __shared__ float s_ea[4][16 * 68];
  __shared__ float s_hid[4][16 * 36];
  __shared__ float s_w[4][16 * 36];
  int wid = threadIdx.x >> 5, lane = threadIdx.x & 31;
  int tile = blockIdx.x * 4 + wid;
  size_t ebase = (size_t)tile * 16;
  stage16x64_async(ea + ebase * 64, s_ea[wid], lane);
  async_wait();
  edge_mlp_tile(s_ea[wid], s_hid[wid], s_w[wid], wv1b, wv2b, wv1_bias, wv2_bias, lane);
  const float* w_s = s_w[wid];
  #pragma unroll
  for (int rep = 0; rep < 2; ++rep) {
    int idx = rep * 32 + lane; int el = idx & 15, h = idx >> 4;
    size_t e = ebase + el;
    int r = eidx[e], cc = eidx[(size_t)Ee + e];
    float alpha = ex_ws[e * 4 + h] / denom[(size_t)r * 4 + h];
    const float* vp = h_v + (size_t)cc * 128 + h * 32;
    float* tp = t_acc + (size_t)r * 128 + h * 32;
    #pragma unroll
    for (int k = 0; k < 32; ++k) atomicAdd(tp + k, alpha * w_s[el * 36 + k] * vp[k]);
  }
}

// ---------------- node pass 2: wvl WMMA + residual + LN2 + ssp + out GEMM -----
__global__ void node_pass2(const float* x, const float* t_acc, const float* denom,
                           const half_t* wvlb, const float* wvl_bias,
                           const float* ln2_g, const float* ln2_b,
                           const half_t* outwb, const float* out_bias, float* out) {
  __shared__ float s_buf[4][16 * 132];
  __shared__ float s_mean[4][16], s_rstd[4][16];
  int wid = threadIdx.x >> 5, lane = threadIdx.x & 31;
  int tile = blockIdx.x * 4 + wid;
  if (tile >= NTILE) return;                  // wave-uniform
  float* buf = s_buf[wid];
  size_t base = (size_t)tile * 16;
  stage16x128_async(t_acc + base * 128, buf, lane);
  async_wait();
  int m0 = (lane >> 4) << 3, ncol = lane & 15;
  // aggr = t @ wvl^T + 1{has edges} * wvl_b, in place per head
  for (int h = 0; h < 4; ++h) {
    v16h a = load_a_lds(buf + h * 32, 132, lane);
    #pragma unroll
    for (int ns = 0; ns < 2; ++ns) {
      v8f c;
      #pragma unroll
      for (int j = 0; j < 8; ++j) c[j] = 0.f;
      c = wmma_f16(a, load_b_blob(wvlb + ns * 512, lane), c);
      float bias = wvl_bias[ns * 16 + ncol];
      #pragma unroll
      for (int j = 0; j < 8; ++j) {
        int m = j + m0; size_t node = base + m;
        float ind = (denom[node * 4 + h] > 0.f) ? 1.f : 0.f;
        buf[m * 132 + h * 32 + ns * 16 + ncol] = c[j] + ind * bias;
      }
    }
  }
  // x2 = aggr + x
  for (int t = lane; t < 2048; t += 32) {
    int r = t >> 7, c = t & 127;
    buf[r * 132 + c] += x[(base + r) * 128 + c];
  }
  if (lane < 16) {
    float s = 0.f, s2 = 0.f;
    for (int c = 0; c < 128; ++c) { float v = buf[lane * 132 + c]; s += v; s2 += v * v; }
    float m = s * (1.f / 128.f);
    float var = s2 * (1.f / 128.f) - m * m;
    s_mean[wid][lane] = m; s_rstd[wid][lane] = rsqrtf(var + 1e-5f);
  }
  // out = ssp(ln2(x2)) @ out_w^T + out_b + x2  (A built on the fly from LDS)
  int row = lane & 15, hi = (lane >> 4) << 3;
  float mm = 0.f, rs = 0.f;
  for (int s = 0; s < 8; ++s) {
    float bias = out_bias[s * 16 + ncol];
    v8f c;
    #pragma unroll
    for (int j = 0; j < 8; ++j) c[j] = bias;
    #pragma unroll
    for (int kc = 0; kc < 4; ++kc) {
      mm = s_mean[wid][row]; rs = s_rstd[wid][row];
      v16h a;
      #pragma unroll
      for (int i = 0; i < 16; ++i) {
        int k = kc * 32 + hi + (i & 7) + ((i >> 3) << 4);
        float yv = (buf[row * 132 + k] - mm) * rs * ln2_g[k] + ln2_b[k];
        a[i] = (half_t)sspf(yv);
      }
      c = wmma_f16(a, load_b_blob(outwb + (s * 4 + kc) * 512, lane), c);
    }
    #pragma unroll
    for (int j = 0; j < 8; ++j) {
      int m = j + m0; size_t node = base + m; int col = s * 16 + ncol;
      out[node * 128 + col] = c[j] + buf[m * 132 + col];
    }
  }
}

extern "C" void kernel_launch(void* const* d_in, const int* in_sizes, int n_in,
                              void* d_out, int out_size, void* d_ws, size_t ws_size,
                              hipStream_t stream) {
  (void)in_sizes; (void)n_in; (void)out_size; (void)ws_size;
  const float* x     = (const float*)d_in[0];
  const float* ea    = (const float*)d_in[1];
  const int*   eidx  = (const int*)  d_in[2];
  const float* k_w   = (const float*)d_in[3];
  const float* q_w   = (const float*)d_in[4];
  const float* v_w   = (const float*)d_in[5];
  const float* wk1_w = (const float*)d_in[6];
  const float* wk1_b = (const float*)d_in[7];
  const float* wk2_w = (const float*)d_in[8];
  const float* wk2_b = (const float*)d_in[9];
  const float* wkl_w = (const float*)d_in[10];
  const float* wkl_b = (const float*)d_in[11];
  const float* wv1_w = (const float*)d_in[12];
  const float* wv1_b = (const float*)d_in[13];
  const float* wv2_w = (const float*)d_in[14];
  const float* wv2_b = (const float*)d_in[15];
  const float* wvl_w = (const float*)d_in[16];
  const float* wvl_b = (const float*)d_in[17];
  const float* out_w = (const float*)d_in[18];
  const float* out_b = (const float*)d_in[19];
  const float* ln1_g = (const float*)d_in[20];
  const float* ln1_b = (const float*)d_in[21];
  const float* ln2_g = (const float*)d_in[22];
  const float* ln2_b = (const float*)d_in[23];
  float* out = (float*)d_out;

  // Workspace layout (~125 MB). Node-level tensors stay L2-resident (192 MB L2).
  char* w = (char*)d_ws; size_t o = 0;
  auto take = [&](size_t bytes) -> char* {
    char* p = w + o; o = (o + bytes + 255) & ~(size_t)255; return p;
  };
  float*    h_k    = (float*)   take((size_t)Nn * 128 * 4);
  float*    h_v    = (float*)   take((size_t)Nn * 128 * 4);
  float*    qt_arr = (float*)   take((size_t)Nn * 128 * 4);
  float*    qb     = (float*)   take((size_t)Nn * 4 * 4);
  unsigned* m_ord  = (unsigned*)take((size_t)Nn * 4 * 4);
  float*    denom  = (float*)   take((size_t)Nn * 4 * 4);
  float*    t_acc  = (float*)   take((size_t)Nn * 128 * 4);
  float*    qk_ws  = (float*)   take((size_t)Ee * 4 * 4);
  float*    ex_ws  = (float*)   take((size_t)Ee * 4 * 4);
  float*    qtw    = (float*)   take((size_t)4 * 32 * 32 * 4);
  float*    u_vec  = (float*)   take((size_t)128 * 4);
  half_t*   blobs  = (half_t*)  take((size_t)BLOB_HALFS * 2);

  hipMemsetAsync(t_acc, 0, (size_t)Nn * 128 * 4, stream);
  init_kernel<<<(Nn * 4 + 255) / 256, 256, 0, stream>>>(m_ord, denom);
  prep_kernel<<<1, 256, 0, stream>>>(k_w, q_w, v_w, wkl_w, wkl_b, wk1_w, wk2_w,
                                     wv1_w, wv2_w, wvl_w, out_w, qtw, u_vec, blobs);
  node_pass1<<<(NTILE + 3) / 4, 128, 0, stream>>>(x, ln1_g, ln1_b,
      blobs + OFF_KW, blobs + OFF_VW, blobs + OFF_QT, u_vec, h_k, h_v, qt_arr, qb);
  edge_pass_a<<<ETILE / 4, 128, 0, stream>>>(ea, eidx, blobs + OFF_WK1, blobs + OFF_WK2,
      wk1_b, wk2_b, qt_arr, h_k, qb, qk_ws, m_ord);
  edge_pass_b<<<(Ee * 4 + 255) / 256, 256, 0, stream>>>(eidx, qk_ws, m_ord, ex_ws, denom);
  edge_pass_c<<<ETILE / 4, 128, 0, stream>>>(ea, eidx, blobs + OFF_WV1, blobs + OFF_WV2,
      wv1_b, wv2_b, h_v, ex_ws, denom, t_acc);
  node_pass2<<<(NTILE + 3) / 4, 128, 0, stream>>>(x, t_acc, denom, blobs + OFF_WVL, wvl_b,
      ln2_g, ln2_b, blobs + OFF_OUT, out_b, out);
}
